// CausalSelfAttn_43525198577893
// MI455X (gfx1250) — compile-verified, matrix operand order checked by reading, not run on hardware
//
#include <hip/hip_runtime.h>

typedef __attribute__((ext_vector_type(16))) _Float16 v16h;
typedef __attribute__((ext_vector_type(8)))  _Float16 v8h;
typedef __attribute__((ext_vector_type(4)))  _Float16 v4h;
typedef __attribute__((ext_vector_type(8)))  float    v8f;
typedef __attribute__((ext_vector_type(4)))  float    v4f;

#define BATCH   4
#define SEQ     2048
#define DMODEL  1024
#define NHEADS  16
#define HDIM    64
#define NEG_BIG (-1e30f)

// Load a 16x32 f16 WMMA fragment (A-layout) from an LDS tile with row stride
// `ld` (elements). Per ISA 7.12.2: lanes 0-15 hold row m=lane, K=0..7 (v0-3)
// and K=16..23 (v4-7); lanes 16-31 hold K=8..15 and K=24..31. The B-fragment
// (32x16) is the mirror with n=lane, so the same loader works on a tile
// stored as [n][k].
__device__ __forceinline__ v16h ldsFrag(const _Float16* tile, int ld) {
    const int lane = threadIdx.x & 31;
    const _Float16* p = tile + (lane & 15) * ld + ((lane >> 4) << 3);
    v8h lo = *(const v8h*)(p);
    v8h hi = *(const v8h*)(p + 16);
    return __builtin_shufflevector(lo, hi, 0,1,2,3,4,5,6,7,8,9,10,11,12,13,14,15);
}

__device__ __forceinline__ v8f wmma16(v16h a, v16h b, v8f c) {
    return __builtin_amdgcn_wmma_f32_16x16x32_f16(false, a, false, b,
                                                  (short)0, c, false, false);
}

// CDNA5 async global->LDS copy (16B per lane), tracked with ASYNCcnt.
// LDS operand is the low 32 bits of the generic shared-memory pointer
// (aperture rule).
__device__ __forceinline__ void async_ld_b128(const void* gptr, void* lptr) {
    unsigned lds = (unsigned)(uintptr_t)lptr;
    asm volatile("global_load_async_to_lds_b128 %0, %1, off"
                 :: "v"(lds), "v"(gptr) : "memory");
}
__device__ __forceinline__ void wait_async() {
    asm volatile("s_wait_asynccnt 0x0" ::: "memory");
}

// ---------------------------------------------------------------------------
// Kernel 1: fused QKV projection.  out = x @ {Wq,Wk,Wv}, f16 results stored
// as [b, h, n, hd]; Q pre-scaled by 1/sqrt(hd).
// grid = (8192/64, 1024/64), block = 128 (4 waves).
// ---------------------------------------------------------------------------
__global__ __launch_bounds__(128) void qkv_kernel(
    const float* __restrict__ x,  const float* __restrict__ Wq,
    const float* __restrict__ Wk, const float* __restrict__ Wv,
    _Float16* __restrict__ qo, _Float16* __restrict__ ko,
    _Float16* __restrict__ vo)
{
    __shared__ __align__(16) _Float16 Xs[64 * 32];      // [m][k]
    __shared__ __align__(16) _Float16 Ws[3][64 * 32];   // [n][k] (transposed)

    const int tid  = threadIdx.x;
    const int wv   = tid >> 5;
    const int lane = tid & 31;
    const int hi   = lane >> 4;
    const int nlo  = lane & 15;
    const int m0   = blockIdx.x * 64;
    const int n0   = blockIdx.y * 64;
    const float* Wmat[3] = {Wq, Wk, Wv};

    const v8f z8 = {0.f,0.f,0.f,0.f,0.f,0.f,0.f,0.f};
    v8f acc[12];
    #pragma unroll
    for (int i = 0; i < 12; ++i) acc[i] = z8;

    for (int kc = 0; kc < DMODEL; kc += 32) {
        // Stage x tile [64][32], fp32 -> f16 (conversion forces VGPR path).
        for (int i = tid; i < 512; i += 128) {
            int r = i >> 3, c = (i & 7) << 2;
            v4f f = *(const v4f*)(x + (size_t)(m0 + r) * DMODEL + kc + c);
            v4h hh = {(_Float16)f[0], (_Float16)f[1], (_Float16)f[2], (_Float16)f[3]};
            *(v4h*)&Xs[r * 32 + c] = hh;
        }
        // Stage W tiles transposed: W[kc..kc+31][n0..n0+63] -> Ws[n][k].
        for (int i = tid; i < 512; i += 128) {
            int kr = i >> 4, c = (i & 15) << 2;
            #pragma unroll
            for (int m = 0; m < 3; ++m) {
                v4f f = *(const v4f*)(Wmat[m] + (size_t)(kc + kr) * DMODEL + n0 + c);
                Ws[m][(c + 0) * 32 + kr] = (_Float16)f[0];
                Ws[m][(c + 1) * 32 + kr] = (_Float16)f[1];
                Ws[m][(c + 2) * 32 + kr] = (_Float16)f[2];
                Ws[m][(c + 3) * 32 + kr] = (_Float16)f[3];
            }
        }
        if (kc + 32 < DMODEL) {
            // CDNA5 global_prefetch_b8 of the next x tile.
            __builtin_prefetch(x + (size_t)(m0 + (tid >> 1)) * DMODEL +
                               (kc + 32) + ((tid & 1) << 4), 0, 1);
        }
        __syncthreads();

        // Preload A + all 12 B fragments, then an uninterrupted WMMA chain.
        v16h a = ldsFrag(Xs + (wv * 16) * 32, 32);
        v16h bf[12];
        #pragma unroll
        for (int m = 0; m < 3; ++m)
            #pragma unroll
            for (int ns = 0; ns < 4; ++ns)
                bf[m * 4 + ns] = ldsFrag(Ws[m] + (ns * 16) * 32, 32);
        #pragma unroll
        for (int i = 0; i < 12; ++i)
            acc[i] = wmma16(a, bf[i], acc[i]);
        __syncthreads();
    }

    _Float16* outp[3] = {qo, ko, vo};
    #pragma unroll
    for (int m = 0; m < 3; ++m) {
        const float scale = (m == 0) ? 0.125f : 1.0f;   // 1/sqrt(64) folded into Q
        #pragma unroll
        for (int ns = 0; ns < 4; ++ns) {
            int col  = n0 + ns * 16 + nlo;
            int head = col >> 6, hc = col & 63;
            #pragma unroll
            for (int r = 0; r < 8; ++r) {
                int gm = m0 + wv * 16 + r + (hi << 3);
                int bb = gm >> 11, ss = gm & (SEQ - 1);
                size_t addr = (((size_t)bb * NHEADS + head) * SEQ + ss) * HDIM + hc;
                outp[m][addr] = (_Float16)(acc[m * 4 + ns][r] * scale);
            }
        }
    }
}

// ---------------------------------------------------------------------------
// Kernel 2: flash attention with online softmax.
// grid = (2048/64, 16, 4); block = 128.  Each wave owns 16 q-rows; WG shares
// 32-key K/V tiles in LDS. ctx written as f16 [b, n, d].
// ---------------------------------------------------------------------------
__global__ __launch_bounds__(128) void attn_kernel(
    const _Float16* __restrict__ q, const _Float16* __restrict__ k,
    const _Float16* __restrict__ v, _Float16* __restrict__ ctx)
{
    __shared__ __align__(16) _Float16 Ks[32 * 64];      // [key][hd] == [n][k] for QK^T
    __shared__ __align__(16) _Float16 Vs[64 * 32];      // [hd][key] == [n][k] for PV
    __shared__ __align__(16) _Float16 Ps[4][16 * 32];   // per-wave P staging

    const int tid  = threadIdx.x;
    const int wv   = tid >> 5;
    const int lane = tid & 31;
    const int hi   = lane >> 4;
    const int nlo  = lane & 15;
    const int q0   = blockIdx.x * 64;
    const int h    = blockIdx.y;
    const int b    = blockIdx.z;
    const size_t headoff = ((size_t)b * NHEADS + h) * SEQ * HDIM;
    const _Float16* qh = q + headoff;
    const _Float16* kh = k + headoff;
    const _Float16* vh = v + headoff;

    // Q fragments for this wave's 16 rows: hd chunks [0,32) and [32,64).
    v16h aq[2];
    #pragma unroll
    for (int kc = 0; kc < 2; ++kc) {
        const _Float16* p = qh + (size_t)(q0 + wv * 16 + nlo) * HDIM + kc * 32 + (hi << 3);
        v8h lo  = *(const v8h*)p;
        v8h hi8 = *(const v8h*)(p + 16);
        aq[kc] = __builtin_shufflevector(lo, hi8, 0,1,2,3,4,5,6,7,8,9,10,11,12,13,14,15);
    }

    const v8f z8 = {0.f,0.f,0.f,0.f,0.f,0.f,0.f,0.f};
    float m_i[8], l_i[8];
    v8f acc[4];
    #pragma unroll
    for (int r = 0; r < 8; ++r) { m_i[r] = NEG_BIG; l_i[r] = 0.f; }
    #pragma unroll
    for (int j = 0; j < 4; ++j) acc[j] = z8;

    for (int kb = 0; kb < q0 + 64; kb += 32) {
        // K tile: straight f16 copy -> CDNA5 async global->LDS (no VGPR trip).
        // V tile: needs transpose -> VGPR path with scalar transposed stores.
        for (int i = tid; i < 256; i += 128) {
            int r = i >> 3, c = (i & 7) << 3;
            async_ld_b128(kh + (size_t)(kb + r) * HDIM + c, &Ks[r * 64 + c]);
            v8h dv = *(const v8h*)(vh + (size_t)(kb + r) * HDIM + c);
            #pragma unroll
            for (int j = 0; j < 8; ++j) Vs[(c + j) * 32 + r] = dv[j];
        }
        wait_async();
        __syncthreads();

        // S = Q K^T for 32 keys: preload 4 B frags, then 4 back-to-back WMMAs.
        v16h bk[4];
        #pragma unroll
        for (int kc = 0; kc < 2; ++kc) {
            bk[kc * 2 + 0] = ldsFrag(Ks + kc * 32, 64);
            bk[kc * 2 + 1] = ldsFrag(Ks + 16 * 64 + kc * 32, 64);
        }
        v8f c0 = z8, c1 = z8;
        #pragma unroll
        for (int kc = 0; kc < 2; ++kc) {
            c0 = wmma16(aq[kc], bk[kc * 2 + 0], c0);
            c1 = wmma16(aq[kc], bk[kc * 2 + 1], c1);
        }
        // Causal mask (scale already folded into Q).
        #pragma unroll
        for (int r = 0; r < 8; ++r) {
            int qr = q0 + wv * 16 + r + (hi << 3);
            if (kb + nlo > qr)      c0[r] = NEG_BIG;
            if (kb + 16 + nlo > qr) c1[r] = NEG_BIG;
        }
        // Online softmax: rows live in 16-lane groups -> xor reductions 1,2,4,8.
        #pragma unroll
        for (int r = 0; r < 8; ++r) {
            float t = fmaxf(c0[r], c1[r]);
            t = fmaxf(t, __shfl_xor(t, 1, 32));
            t = fmaxf(t, __shfl_xor(t, 2, 32));
            t = fmaxf(t, __shfl_xor(t, 4, 32));
            t = fmaxf(t, __shfl_xor(t, 8, 32));
            float mn    = fmaxf(m_i[r], t);
            float alpha = __expf(m_i[r] - mn);
            float p0    = __expf(c0[r] - mn);
            float p1    = __expf(c1[r] - mn);
            float rs = p0 + p1;
            rs += __shfl_xor(rs, 1, 32);
            rs += __shfl_xor(rs, 2, 32);
            rs += __shfl_xor(rs, 4, 32);
            rs += __shfl_xor(rs, 8, 32);
            l_i[r] = l_i[r] * alpha + rs;
            m_i[r] = mn;
            #pragma unroll
            for (int j = 0; j < 4; ++j) acc[j][r] *= alpha;
            // C-layout -> A-layout bounce through LDS.
            Ps[wv][(r + (hi << 3)) * 32 + nlo]      = (_Float16)p0;
            Ps[wv][(r + (hi << 3)) * 32 + 16 + nlo] = (_Float16)p1;
        }
        __builtin_amdgcn_wave_barrier();   // keep P stores ahead of frag loads

        // PV: preload P frag + 4 V frags, then 4 back-to-back WMMAs.
        v16h ap = ldsFrag(Ps[wv], 32);
        v16h bv[4];
        #pragma unroll
        for (int j = 0; j < 4; ++j) bv[j] = ldsFrag(Vs + (j * 16) * 32, 32);
        #pragma unroll
        for (int j = 0; j < 4; ++j) acc[j] = wmma16(ap, bv[j], acc[j]);
        __syncthreads();
    }

    // Normalize and store ctx as f16 [b, seq, d].
    #pragma unroll
    for (int j = 0; j < 4; ++j) {
        int col = h * HDIM + j * 16 + nlo;
        #pragma unroll
        for (int r = 0; r < 8; ++r) {
            int qr = q0 + wv * 16 + r + (hi << 3);
            ctx[((size_t)b * SEQ + qr) * DMODEL + col] = (_Float16)(acc[j][r] / l_i[r]);
        }
    }
}

// ---------------------------------------------------------------------------
// Kernel 3: output projection.  out(f32) = ctx(f16) @ Wo + bo.
// grid = (8192/64, 1024/64), block = 128.
// ---------------------------------------------------------------------------
__global__ __launch_bounds__(128) void out_kernel(
    const _Float16* __restrict__ ctx, const float* __restrict__ Wo,
    const float* __restrict__ bo, float* __restrict__ out)
{
    __shared__ __align__(16) _Float16 Cs[64 * 32];
    __shared__ __align__(16) _Float16 Ws[64 * 32];

    const int tid  = threadIdx.x;
    const int wv   = tid >> 5;
    const int lane = tid & 31;
    const int hi   = lane >> 4;
    const int nlo  = lane & 15;
    const int m0   = blockIdx.x * 64;
    const int n0   = blockIdx.y * 64;

    const v8f z8 = {0.f,0.f,0.f,0.f,0.f,0.f,0.f,0.f};
    v8f acc[4];
    #pragma unroll
    for (int ns = 0; ns < 4; ++ns) acc[ns] = z8;

    for (int kc = 0; kc < DMODEL; kc += 32) {
        // ctx tile: straight f16 copy -> async global->LDS.
        for (int i = tid; i < 256; i += 128) {
            int r = i >> 2, c = (i & 3) << 3;
            async_ld_b128(ctx + (size_t)(m0 + r) * DMODEL + kc + c, &Cs[r * 32 + c]);
        }
        // Wo tile: f32 -> f16 + transpose, VGPR path.
        for (int i = tid; i < 512; i += 128) {
            int kr = i >> 4, c = (i & 15) << 2;
            v4f f = *(const v4f*)(Wo + (size_t)(kc + kr) * DMODEL + n0 + c);
            Ws[(c + 0) * 32 + kr] = (_Float16)f[0];
            Ws[(c + 1) * 32 + kr] = (_Float16)f[1];
            Ws[(c + 2) * 32 + kr] = (_Float16)f[2];
            Ws[(c + 3) * 32 + kr] = (_Float16)f[3];
        }
        if (kc + 32 < DMODEL) {
            __builtin_prefetch(ctx + (size_t)(m0 + (tid >> 1)) * DMODEL +
                               (kc + 32) + ((tid & 1) << 4), 0, 1);
        }
        wait_async();
        __syncthreads();

        v16h a = ldsFrag(Cs + (wv * 16) * 32, 32);
        v16h bf[4];
        #pragma unroll
        for (int ns = 0; ns < 4; ++ns) bf[ns] = ldsFrag(Ws + (ns * 16) * 32, 32);
        #pragma unroll
        for (int ns = 0; ns < 4; ++ns) acc[ns] = wmma16(a, bf[ns], acc[ns]);
        __syncthreads();
    }

    #pragma unroll
    for (int ns = 0; ns < 4; ++ns) {
        int col = n0 + ns * 16 + nlo;
        float bias = bo[col];
        #pragma unroll
        for (int r = 0; r < 8; ++r) {
            int gm = m0 + wv * 16 + r + (hi << 3);
            out[(size_t)gm * DMODEL + col] = acc[ns][r] + bias;
        }
    }
}

// ---------------------------------------------------------------------------
extern "C" void kernel_launch(void* const* d_in, const int* in_sizes, int n_in,
                              void* d_out, int out_size, void* d_ws, size_t ws_size,
                              hipStream_t stream) {
    (void)in_sizes; (void)n_in; (void)out_size; (void)ws_size;
    const float* x  = (const float*)d_in[0];
    const float* Wq = (const float*)d_in[1];
    const float* Wk = (const float*)d_in[2];
    const float* Wv = (const float*)d_in[3];
    const float* Wo = (const float*)d_in[4];
    const float* bo = (const float*)d_in[5];
    float* out = (float*)d_out;

    const size_t EN = (size_t)BATCH * SEQ * DMODEL;   // 8,388,608 elements
    _Float16* qws = (_Float16*)d_ws;                  // [b,h,n,hd] f16
    _Float16* kws = qws + EN;
    _Float16* vws = kws + EN;
    _Float16* cws = vws + EN;                         // ctx [b,n,d] f16

    qkv_kernel<<<dim3((BATCH * SEQ) / 64, DMODEL / 64), 128, 0, stream>>>(
        x, Wq, Wk, Wv, qws, kws, vws);
    attn_kernel<<<dim3(SEQ / 64, NHEADS, BATCH), 128, 0, stream>>>(
        qws, kws, vws, cws);
    out_kernel<<<dim3((BATCH * SEQ) / 64, DMODEL / 64), 128, 0, stream>>>(
        cws, Wo, bo, out);
}
